// BConv_34024730919494
// MI455X (gfx1250) — 8-bit matrix operand order repaired and checked by reading, compiled, not run on hardware
//
#include <hip/hip_runtime.h>
#include <stdint.h>


typedef __attribute__((ext_vector_type(8))) int   v8i;
typedef __attribute__((ext_vector_type(4))) int   v4i;
typedef __attribute__((ext_vector_type(2))) int   v2i;
typedef __attribute__((ext_vector_type(4))) float v4f;

#define HH 112
#define WW 112
#define CC_IN 128
#define KK 128
#define NB 32

__device__ __forceinline__ signed char sgn8(float v) {
    return (signed char)((v > 0.0f) ? 1 : ((v < 0.0f) ? -1 : 0));
}

__global__ void __launch_bounds__(256) pack_x_kernel(const float* __restrict__ x,
                                                     signed char* __restrict__ xq) {
    __shared__ __align__(16) signed char tile[WW * 132];
    const int h = blockIdx.x;
    const int n = blockIdx.y;
    const int tid = threadIdx.x;

    const size_t nbase = (size_t)n * CC_IN * (HH * WW) + (size_t)h * WW;
    for (int i = tid; i < CC_IN * WW; i += 256) {
        int c = i / WW;
        int w = i - c * WW;
        float v = x[nbase + (size_t)c * (HH * WW) + w];
        tile[w * 132 + c] = sgn8(v);
    }
    __syncthreads();

    unsigned int* dst = (unsigned int*)(xq + ((size_t)n * HH + h) * (size_t)(WW * CC_IN));
    for (int j = tid; j < (WW * CC_IN) / 4; j += 256) {
        int w  = j >> 5;
        int c4 = j & 31;
        dst[j] = *(const unsigned int*)(tile + w * 132 + c4 * 4);
    }
}

__global__ void __launch_bounds__(256) pack_w_kernel(const float* __restrict__ wsrc,
                                                     signed char* __restrict__ wq) {
    int idx = blockIdx.x * blockDim.x + threadIdx.x;
    if (idx >= KK * CC_IN * 9) return;
    int kw = idx % 3;
    int t  = idx / 3;
    int kh = t % 3;
    t /= 3;
    int c = t % CC_IN;
    int k = t / CC_IN;
    wq[((kh * 3 + kw) * KK + k) * CC_IN + c] = sgn8(wsrc[idx]);
}

__global__ void __launch_bounds__(256, 4)
bconv_wmma_kernel(const signed char* __restrict__ xq,
                  const signed char* __restrict__ wq,
                  float* __restrict__ out) {
    __shared__ __align__(16) unsigned char smem[8 * 16 * 20 * 4];
    signed char* sx = (signed char*)smem;
    float*       st = (float*)smem;

    const int ow0 = blockIdx.x * 16;
    const int oh  = blockIdx.y;
    const int n   = blockIdx.z;
    const int tid = threadIdx.x;

    unsigned int* s32 = (unsigned int*)sx;
    const int ndw = (3 * 18 * CC_IN) / 4;
    for (int i = tid; i < ndw; i += 256) {
        int c4  = i & 31;
        int pix = i >> 5;
        int iw  = pix % 18;
        int ih  = pix / 18;
        int h = oh - 1 + ih;
        int w = ow0 - 1 + iw;
        unsigned int v = 0u;
        if ((unsigned)h < (unsigned)HH && (unsigned)w < (unsigned)WW) {
            v = *(const unsigned int*)(xq + (((size_t)n * HH + h) * WW + w) * CC_IN + c4 * 4);
        }
        s32[i] = v;
    }
    __syncthreads();

    const int wave = tid >> 5;
    const int lane = tid & 31;
    const int m    = lane & 15;
    const int half = lane >> 4;
    const int kbase = wave * 16;

    v8i acc = {};

#pragma unroll
    for (int kh = 0; kh < 3; ++kh) {
#pragma unroll
        for (int kw = 0; kw < 3; ++kw) {
            const signed char* arow = sx + (kh * 18 + (m + kw)) * CC_IN;
            const signed char* brow = wq + ((size_t)(kh * 3 + kw) * KK + (kbase + m)) * CC_IN;
#pragma unroll
            for (int cc = 0; cc < 2; ++cc) {
                const signed char* ab = arow + cc * 64 + half * 8;
                v2i a01 = *(const v2i*)(ab);
                v2i a23 = *(const v2i*)(ab + 16);
                v2i a45 = *(const v2i*)(ab + 32);
                v2i a67 = *(const v2i*)(ab + 48);
                v8i a = {a01.x, a01.y, a23.x, a23.y, a45.x, a45.y, a67.x, a67.y};

                const signed char* bb = brow + cc * 64 + half * 8;
                v2i b01 = *(const v2i*)(bb);
                v2i b23 = *(const v2i*)(bb + 16);
                v2i b45 = *(const v2i*)(bb + 32);
                v2i b67 = *(const v2i*)(bb + 48);
                v8i b = {b01.x, b01.y, b23.x, b23.y, b45.x, b45.y, b67.x, b67.y};

                acc = __builtin_amdgcn_wmma_i32_16x16x64_iu8(
                     true, a,  true, b, acc,
                     false,  false);
            }
        }
    }

    __syncthreads();

    {
        const int ncol = lane & 15;
        float* row = st + (wave * 16 + ncol) * 20 + half * 8;
        v4f f0 = {(float)acc[0], (float)acc[1], (float)acc[2], (float)acc[3]};
        v4f f1 = {(float)acc[4], (float)acc[5], (float)acc[6], (float)acc[7]};
        *(v4f*)(row)     = f0;
        *(v4f*)(row + 4) = f1;
    }
    __syncthreads();

    {
        float* obase = out + (((size_t)n * KK) * HH + oh) * WW + ow0;
        for (int j = tid; j < 512; j += 256) {
            int k  = j >> 2;
            int p0 = (j & 3) * 4;
            v4f v = *(const v4f*)(st + k * 20 + p0);
            *(v4f*)(obase + (size_t)k * (HH * WW) + p0) = v;
        }
    }
}

extern "C" void kernel_launch(void* const* d_in, const int* in_sizes, int n_in,
                              void* d_out, int out_size, void* d_ws, size_t ws_size,
                              hipStream_t stream) {
    const float* x    = (const float*)d_in[0];
    const float* wsrc = (const float*)d_in[1];
    float* out        = (float*)d_out;

    signed char* xq = (signed char*)d_ws;
    signed char* wq = xq + (size_t)NB * HH * WW * CC_IN;

    dim3 gx(HH, NB);
    pack_x_kernel<<<gx, 256, 0, stream>>>(x, xq);

    pack_w_kernel<<<(KK * CC_IN * 9 + 255) / 256, 256, 0, stream>>>(wsrc, wq);

    dim3 gc(WW / 16, HH, NB);
    bconv_wmma_kernel<<<gc, 256, 0, stream>>>(xq, wq, out);
}
